// Attention_45749991637079
// MI455X (gfx1250) — matrix operand order repaired and checked by reading, not compiled, not run on hardware
//
#include <hip/hip_runtime.h>
#include <hip/hip_bf16.h>
#include <math.h>

// ---------- types ----------
typedef __attribute__((ext_vector_type(16))) __bf16 v16bf;
typedef __attribute__((ext_vector_type(2)))  __bf16 v2bf;
typedef __attribute__((ext_vector_type(8)))  float  v8f;
typedef __attribute__((ext_vector_type(4)))  unsigned int u32x4;
typedef __attribute__((ext_vector_type(2)))  unsigned int u32x2;
typedef __attribute__((ext_vector_type(4)))  float  f32x4;

union BfPack { u32x4 q[2]; v16bf v; };

// Native float->bf16 conversion (lowers to hardware v_cvt_*bf16_f32, RNE).
__device__ __forceinline__ unsigned short f2bf(float f) {
  __bf16 h = (__bf16)f;
  return __builtin_bit_cast(unsigned short, h);
}
__device__ __forceinline__ float bf2f(unsigned short v) {
  return __uint_as_float(((unsigned)v) << 16);
}
__device__ __forceinline__ unsigned pack2(float a, float b) {
  v2bf p;
  p[0] = (__bf16)a;
  p[1] = (__bf16)b;
  return __builtin_bit_cast(unsigned, p);
}

// ---------- K0: pack w_qkv rows {0..191, 384..575} to bf16 ----------
__global__ void prep_a1_kernel(const float* __restrict__ w_qkv,
                               unsigned short* __restrict__ A1) {
  const int i = blockIdx.x * 256 + threadIdx.x;
  if (i >= 384 * 192) return;
  const int r = i / 192, cc = i - r * 192;
  const int o = (r < 192) ? r : (r + 192);       // skip dead k-rows
  A1[i] = f2bf(w_qkv[(size_t)o * 192 + cc]);
}

// ---------- K1/K5: bf16 WMMA GEMM  C[b] = A[b](Mx192) * B[b](192x16384) ----------
// Block: 256 thr = 8 waves (2x4).  Block tile 64(M) x 256(N).  Wave tile 32x64.
// LDS B tile staged as [n][k] bf16, row stride 40 (80B, 16B aligned, conflict-free).
template<bool BBF16, bool COUT_BF16>
__global__ __launch_bounds__(256)
void gemm_wmma_bf16(const unsigned short* __restrict__ A, size_t aBatch,
                    const void* __restrict__ Bv, size_t bBatch,
                    void* __restrict__ C, size_t cBatch) {
  constexpr int N = 16384;
  constexpr int K = 192;
  __shared__ __attribute__((aligned(16))) unsigned short smem[256 * 40];

  const int t       = threadIdx.x;
  const int lane    = t & 31;
  const int lane16  = lane & 15;
  const int halfsel = (lane >> 4) & 1;
  const int wave    = t >> 5;
  const int wave_m  = wave >> 2;   // 0..1
  const int wave_n  = wave & 3;    // 0..3

  const int b  = blockIdx.z;
  const int m0 = blockIdx.y * 64;
  const int n0 = blockIdx.x * 256;

  const unsigned short* Ab = A + aBatch * (size_t)b;

  v8f acc[2][4];
  #pragma unroll
  for (int mi = 0; mi < 2; ++mi)
    #pragma unroll
    for (int ni = 0; ni < 4; ++ni)
      #pragma unroll
      for (int r = 0; r < 8; ++r) acc[mi][ni][r] = 0.0f;

  const int tn      = t & 63;       // 64 n-groups of 4
  const int n_local = tn * 4;
  const int tk      = t >> 6;       // 0..3, each covers 8 k-rows

  for (int kt = 0; kt < K / 32; ++kt) {
    // ---- stage B[32k x 256n] -> LDS [n][k] (bf16) ----
    if (BBF16) {
      const unsigned short* Bb = (const unsigned short*)Bv + bBatch * (size_t)b;
      unsigned r[8][2];
      #pragma unroll
      for (int i = 0; i < 8; ++i) {
        const u32x2 v = *(const u32x2*)(Bb + (size_t)(kt*32 + tk*8 + i) * N + n0 + n_local);
        r[i][0] = v[0]; r[i][1] = v[1];
      }
      #pragma unroll
      for (int j = 0; j < 4; ++j) {
        unsigned e[8];
        #pragma unroll
        for (int i = 0; i < 8; ++i)
          e[i] = (r[i][j >> 1] >> ((j & 1) * 16)) & 0xFFFFu;
        u32x4 out;
        out[0] = e[0] | (e[1] << 16); out[1] = e[2] | (e[3] << 16);
        out[2] = e[4] | (e[5] << 16); out[3] = e[6] | (e[7] << 16);
        *(u32x4*)&smem[(n_local + j) * 40 + tk * 8] = out;
      }
    } else {
      const float* Bb = (const float*)Bv + bBatch * (size_t)b;
      float f[8][4];
      #pragma unroll
      for (int i = 0; i < 8; ++i) {
        const f32x4 v = *(const f32x4*)(Bb + (size_t)(kt*32 + tk*8 + i) * N + n0 + n_local);
        f[i][0] = v[0]; f[i][1] = v[1]; f[i][2] = v[2]; f[i][3] = v[3];
      }
      #pragma unroll
      for (int j = 0; j < 4; ++j) {
        u32x4 out;
        out[0] = pack2(f[0][j], f[1][j]); out[1] = pack2(f[2][j], f[3][j]);
        out[2] = pack2(f[4][j], f[5][j]); out[3] = pack2(f[6][j], f[7][j]);
        *(u32x4*)&smem[(n_local + j) * 40 + tk * 8] = out;
      }
    }
    __syncthreads();

    // ---- A operands: 16x32 bf16, lane half picks K {0..7,16..23} / {8..15,24..31}
    BfPack a[2];
    #pragma unroll
    for (int mi = 0; mi < 2; ++mi) {
      const int m = m0 + wave_m * 32 + mi * 16 + lane16;
      const unsigned short* ap = Ab + (size_t)m * K + kt * 32 + halfsel * 8;
      a[mi].q[0] = *(const u32x4*)ap;
      a[mi].q[1] = *(const u32x4*)(ap + 16);
    }
    // ---- B operands + WMMA: lane half picks K {0..7,16..23} / {8..15,24..31}
    #pragma unroll
    for (int ni = 0; ni < 4; ++ni) {
      BfPack bb;
      const int n = wave_n * 64 + ni * 16 + lane16;
      const unsigned short* bp = &smem[n * 40 + halfsel * 8];
      bb.q[0] = *(const u32x4*)bp;
      bb.q[1] = *(const u32x4*)(bp + 16);
      #pragma unroll
      for (int mi = 0; mi < 2; ++mi) {
        acc[mi][ni] = __builtin_amdgcn_wmma_f32_16x16x32_bf16(
            false, a[mi].v, false, bb.v, (short)0, acc[mi][ni], false, false);
      }
    }
    __syncthreads();
  }

  // ---- epilogue: C/D layout -> VGPR r holds row (halfsel*8 + r), col = lane16
  #pragma unroll
  for (int mi = 0; mi < 2; ++mi) {
    #pragma unroll
    for (int ni = 0; ni < 4; ++ni) {
      const int mt = m0 + wave_m * 32 + mi * 16 + halfsel * 8;
      const int nt = n0 + wave_n * 64 + ni * 16 + lane16;
      if (COUT_BF16) {
        unsigned short* cp = (unsigned short*)C + cBatch * (size_t)b + (size_t)mt * N + nt;
        #pragma unroll
        for (int r = 0; r < 8; ++r) cp[(size_t)r * N] = f2bf(acc[mi][ni][r]);
      } else {
        float* cp = (float*)C + cBatch * (size_t)b + (size_t)mt * N + nt;
        #pragma unroll
        for (int r = 0; r < 8; ++r) cp[(size_t)r * N] = acc[mi][ni][r];
      }
    }
  }
}

// ---------- K2: depthwise 3x3 SAME conv (fp32 accum on bf16 qkv) ----------
// rows 0..191  = q-channels -> conv only at even pixels -> qd (fp32, 64x64)
// rows 192..383= v-channels -> full res -> vbf (bf16)
__global__ __launch_bounds__(256)
void dwconv_kernel(const unsigned short* __restrict__ qkv,
                   const float* __restrict__ w_dw,
                   float* __restrict__ qd,
                   unsigned short* __restrict__ vbf) {
  const int r = blockIdx.y;
  const int b = blockIdx.z;
  const unsigned short* in = qkv + ((size_t)b * 384 + r) * 16384;
  const int o = (r < 192) ? r : (r + 192);   // original qkv channel for w_dw
  float w[9];
  #pragma unroll
  for (int i = 0; i < 9; ++i) w[i] = w_dw[o * 9 + i];

  if (r < 192) {
    if (blockIdx.x >= 16) return;            // only 4096 outputs
    const int idx = blockIdx.x * 256 + threadIdx.x;
    const int yo = idx >> 6, xo = idx & 63;
    const int y = yo * 2, x = xo * 2;
    float s = 0.f;
    #pragma unroll
    for (int ky = 0; ky < 3; ++ky) {
      const int yy = y + ky - 1;
      if (yy < 0 || yy > 127) continue;
      #pragma unroll
      for (int kx = 0; kx < 3; ++kx) {
        const int xx = x + kx - 1;
        if (xx < 0 || xx > 127) continue;
        s += bf2f(in[yy * 128 + xx]) * w[ky * 3 + kx];
      }
    }
    qd[((size_t)b * 192 + r) * 4096 + idx] = s;
  } else {
    const int idx = blockIdx.x * 256 + threadIdx.x;
    const int y = idx >> 7, x = idx & 127;
    float s = 0.f;
    #pragma unroll
    for (int ky = 0; ky < 3; ++ky) {
      const int yy = y + ky - 1;
      if (yy < 0 || yy > 127) continue;
      #pragma unroll
      for (int kx = 0; kx < 3; ++kx) {
        const int xx = x + kx - 1;
        if (xx < 0 || xx > 127) continue;
        s += bf2f(in[yy * 128 + xx]) * w[ky * 3 + kx];
      }
    }
    vbf[((size_t)b * 192 + (r - 192)) * 16384 + idx] = f2bf(s);
  }
}

// ---------- K3: Gram(24x24 over n=4096) -> normalize -> softmax ----------
// One block per (b,head), 576 threads (one per (c,d) pair).
__global__ __launch_bounds__(576)
void gram_softmax_kernel(const float* __restrict__ qd,
                         const float* __restrict__ temperature,
                         float* __restrict__ attn) {
  __shared__ __attribute__((aligned(16))) float chunk[24 * 132]; // padded rows
  __shared__ float G[576];
  const int b = blockIdx.x >> 3;
  const int h = blockIdx.x & 7;
  const int t = threadIdx.x;
  const int c = t / 24, d = t - c * 24;
  const float* base = qd + ((size_t)b * 192 + h * 24) * 4096;

  float acc = 0.f;
  for (int ch = 0; ch < 32; ++ch) {
    for (int i = t; i < 24 * 128; i += 576) {
      const int cc = i >> 7, j = i & 127;
      chunk[cc * 132 + j] = base[(size_t)cc * 4096 + ch * 128 + j];
    }
    __syncthreads();
    const float* pc = chunk + c * 132;
    const float* pd = chunk + d * 132;
    #pragma unroll 4
    for (int j = 0; j < 128; j += 4) {
      const f32x4 a = *(const f32x4*)(pc + j);
      const f32x4 v = *(const f32x4*)(pd + j);
      acc += a[0]*v[0] + a[1]*v[1] + a[2]*v[2] + a[3]*v[3];
    }
    __syncthreads();
  }
  G[c * 24 + d] = acc;
  __syncthreads();

  if (t < 24) {
    const float temp = temperature[h];
    const float nc = sqrtf(G[t * 24 + t]);
    float logit[24];
    float mx = -3.0e38f;
    for (int dd = 0; dd < 24; ++dd) {
      const float nd = sqrtf(G[dd * 24 + dd]);
      const float l = G[t * 24 + dd] / fmaxf(nc * nd, 1e-24f) * temp;
      logit[dd] = l;
      mx = fmaxf(mx, l);
    }
    float sum = 0.f;
    for (int dd = 0; dd < 24; ++dd) { const float e = expf(logit[dd] - mx); logit[dd] = e; sum += e; }
    const float inv = 1.f / sum;
    float* ap = attn + (((size_t)b * 8 + h) * 24 + t) * 24;
    for (int dd = 0; dd < 24; ++dd) ap[dd] = logit[dd] * inv;
  }
}

// ---------- K4: Weff[b] = w_proj * blockdiag(attn[b])  (bf16 out) ----------
__global__ __launch_bounds__(192)
void weff_kernel(const float* __restrict__ w_proj,
                 const float* __restrict__ attn,
                 unsigned short* __restrict__ Weff) {
  __shared__ float at[8 * 576];
  const int b = blockIdx.x;
  const int o = threadIdx.x;
  for (int i = o; i < 4608; i += 192) at[i] = attn[(size_t)b * 4608 + i];
  __syncthreads();
  const float* wp = w_proj + (size_t)o * 192;
  unsigned short* wout = Weff + ((size_t)b * 192 + o) * 192;
  for (int h = 0; h < 8; ++h) {
    float wrow[24];
    #pragma unroll
    for (int cc = 0; cc < 24; ++cc) wrow[cc] = wp[h * 24 + cc];
    for (int dd = 0; dd < 24; ++dd) {
      float s = 0.f;
      #pragma unroll
      for (int cc = 0; cc < 24; ++cc) s += wrow[cc] * at[h * 576 + cc * 24 + dd];
      wout[h * 24 + dd] = f2bf(s);
    }
  }
}

// ---------- host ----------
extern "C" void kernel_launch(void* const* d_in, const int* in_sizes, int n_in,
                              void* d_out, int out_size, void* d_ws, size_t ws_size,
                              hipStream_t stream) {
  (void)in_sizes; (void)n_in; (void)out_size; (void)ws_size;
  const float* x           = (const float*)d_in[0];
  const float* w_qkv       = (const float*)d_in[1];
  const float* w_dw        = (const float*)d_in[2];
  const float* w_proj      = (const float*)d_in[3];
  const float* temperature = (const float*)d_in[4];
  float* out = (float*)d_out;

  char* ws = (char*)d_ws;
  size_t off = 0;
  auto take = [&](size_t bytes) -> char* {
    char* p = ws + off;
    off += (bytes + 255) & ~(size_t)255;
    return p;
  };
  unsigned short* A1     = (unsigned short*)take((size_t)384 * 192 * 2);
  float*          qd     = (float*)take((size_t)8 * 192 * 4096 * 4);     // ~25 MB
  unsigned short* vbf    = (unsigned short*)take((size_t)8 * 192 * 16384 * 2); // ~50 MB
  float*          attn   = (float*)take((size_t)8 * 8 * 24 * 24 * 4);
  unsigned short* Weff   = (unsigned short*)take((size_t)8 * 192 * 192 * 2);
  unsigned short* qkvbuf = (unsigned short*)take((size_t)8 * 384 * 16384 * 2); // ~100 MB

  // K0: weight pack (dead k-rows dropped)
  prep_a1_kernel<<<288, 256, 0, stream>>>(w_qkv, A1);
  // K1: qkv GEMM (q+v rows only), bf16 WMMA, bf16 out
  gemm_wmma_bf16<false, true><<<dim3(64, 6, 8), 256, 0, stream>>>(
      A1, 0, (const void*)x, (size_t)192 * 16384, (void*)qkvbuf, (size_t)384 * 16384);
  // K2: depthwise conv; q at even pixels -> qd, v full-res -> vbf
  dwconv_kernel<<<dim3(64, 384, 8), 256, 0, stream>>>(qkvbuf, w_dw, qd, vbf);
  // K3: Gram + normalize + softmax per (b,head)
  gram_softmax_kernel<<<64, 576, 0, stream>>>(qd, temperature, attn);
  // K4: fold attention into projection
  weff_kernel<<<8, 192, 0, stream>>>(w_proj, attn, Weff);
  // K5: out[b] = Weff[b] x v[b], bf16 WMMA, fp32 out
  gemm_wmma_bf16<true, false><<<dim3(64, 3, 8), 256, 0, stream>>>(
      Weff, (size_t)192 * 192, (const void*)vbf, (size_t)192 * 16384,
      (void*)out, (size_t)192 * 16384);
}
